// KernelRidgeRegression_65481071402881
// MI455X (gfx1250) — compile-verified, matrix operand order checked by reading, not run on hardware
//
#include <hip/hip_runtime.h>

#define GAMMA 1.0f
#define KRR_USE_TDM 1   // flip to 0 to fall back to manual global->reg->LDS staging

// CDNA5 WMMA types (wave32): 16 bf16 elements = 8 VGPRs, 8 f32 acc = 8 VGPRs
typedef __attribute__((ext_vector_type(16))) __bf16        v16bf;
typedef __attribute__((ext_vector_type(8)))  float         v8f;
typedef __attribute__((ext_vector_type(4))) unsigned int   uint4v;
typedef __attribute__((ext_vector_type(8))) unsigned int   uint8v;

union FragU { uint4v q[2]; v16bf v; };

// f32 -> bf16 round-to-nearest-even (fallback path)
__device__ __forceinline__ unsigned short f2bf(float x) {
  unsigned u = __builtin_bit_cast(unsigned, x);
  u += 0x7FFFu + ((u >> 16) & 1u);
  return (unsigned short)(u >> 16);
}

__device__ __forceinline__ unsigned pack_bf16(float a, float b) {
#if __has_builtin(__builtin_amdgcn_cvt_pk_bf16_f32)
  typedef __attribute__((ext_vector_type(2))) __bf16 v2bf;
  v2bf p = __builtin_amdgcn_cvt_pk_bf16_f32(a, b);
  return __builtin_bit_cast(unsigned, p);
#else
  return (unsigned)f2bf(a) | ((unsigned)f2bf(b) << 16);
#endif
}

// Load one 16x32 bf16 WMMA operand fragment from LDS (two ds_load_b128).
__device__ __forceinline__ v16bf load_frag_lds(const unsigned short* base, int stride, int lane) {
  const int row = lane & 15;
  const int kb  = (lane >> 4) << 3;  // 0 or 8
  const unsigned short* p = base + row * stride + kb;
  FragU f;
  f.q[0] = *(const uint4v*)(p);        // K = kb .. kb+7
  f.q[1] = *(const uint4v*)(p + 16);   // K = kb+16 .. kb+23
  return f.v;
}

#if KRR_USE_TDM
// Issue one TDM load of a 512-row x 32-col bf16 tile (rows of train_X) into LDS,
// padded 64B data + 16B pad per row -> LDS row stride 40 bf16 (conflict-free frags).
// D# per CDNA5 ISA; groups 2/3 unused (2D tile) -> zeros.
__device__ __forceinline__ void tdm_load_b_tile(const unsigned short* Tbf, int n0, int kt,
                                                unsigned lds_byte_addr) {
  unsigned long long ga = (unsigned long long)(const void*)Tbf +
                          ((unsigned long long)n0 * 512ull + (unsigned long long)kt * 32ull) * 2ull;
  uint4v g0;
  g0[0] = 1u;                                            // count=1, no gather
  g0[1] = lds_byte_addr;                                 // lds_addr
  g0[2] = (unsigned)ga;                                  // global_addr[31:0]
  g0[3] = ((unsigned)(ga >> 32) & 0x01FFFFFFu) | (2u << 30);  // global_addr[56:32], type=2
  uint8v g1;
  g1[0] = (1u << 16) | (1u << 20) | (3u << 22) | (3u << 25);  // data_size=2B, pad_en, interval=16DW, amount=4DW
  g1[1] = 512u << 16;                                    // tensor_dim0[15:0]=512 (bits 63:48)
  g1[2] = 8192u << 16;                                   // tensor_dim1[15:0]=8192 (bits 95:80)
  g1[3] = 32u << 16;                                     // tile_dim0=32 (bits 127:112)
  g1[4] = 512u;                                          // tile_dim1=512, tile_dim2=0
  g1[5] = 512u;                                          // tensor_dim0_stride[31:0]=512
  g1[6] = 0u;
  g1[7] = 0u;
  uint4v gz;
  gz[0] = 0u; gz[1] = 0u; gz[2] = 0u; gz[3] = 0u;
  asm volatile("tensor_load_to_lds %0, %1, %2, %3"
               :: "s"(g0), "s"(g1), "s"(gz), "s"(gz)
               : "memory");
}
#endif

// ------- Kernel 1: fused f32->bf16 conversion + row norms (one pass over inputs) -------
__global__ __launch_bounds__(256) void krr_prep(const float* __restrict__ X,
                                                const float* __restrict__ T,
                                                unsigned short* __restrict__ Xbf,
                                                unsigned short* __restrict__ Tbf,
                                                float* __restrict__ norms) {
  const int lane = threadIdx.x & 31;
  const int wid  = threadIdx.x >> 5;
  const int row  = blockIdx.x * 8 + wid;   // 0..16383
  const float* src;
  unsigned short* dst;
  if (row < 8192) { src = X + (size_t)row * 512;          dst = Xbf + (size_t)row * 512; }
  else            { src = T + (size_t)(row - 8192) * 512; dst = Tbf + (size_t)(row - 8192) * 512; }
  float s = 0.f;
#pragma unroll
  for (int i = 0; i < 4; ++i) {
    float4 v = ((const float4*)src)[lane + i * 32];
    s += v.x * v.x + v.y * v.y + v.z * v.z + v.w * v.w;
    unsigned lo = pack_bf16(v.x, v.y);
    unsigned hi = pack_bf16(v.z, v.w);
    *(uint2*)(dst + (size_t)(lane + i * 32) * 4) = make_uint2(lo, hi);
  }
#pragma unroll
  for (int m = 16; m >= 1; m >>= 1) s += __shfl_xor(s, m, 32);
  if (lane == 0) norms[row] = s;
}

// ------- Kernel 2: bf16-WMMA GEMM + fused RBF epilogue -------
// Workgroup (256 thr = 8 waves) owns a 32-row strip; A-panel (32x512 bf16) LDS-resident.
// N-loop in 512-col tiles; wave covers 32Mx64N -> 8 WMMAs / K-step. B tiles are
// triple-buffered and staged by the Tensor Data Mover (one instruction per K-step).
__global__ __launch_bounds__(256) void krr_main(const unsigned short* __restrict__ Xbf,
                                                const unsigned short* __restrict__ Tbf,
                                                const float* __restrict__ dual,
                                                const float* __restrict__ norms,
                                                float* __restrict__ out) {
  __shared__ unsigned short As[32 * 520];      // stride 520 -> 16B aligned, conflict-free frags
  __shared__ unsigned short Bs[3][512 * 40];   // stride 40  -> 16B aligned, conflict-free frags
  __shared__ float red[8][32];

  const int tid  = threadIdx.x;
  const int lane = tid & 31;
  const int wid  = tid >> 5;           // 0..7 : 64-col slice of the 512-col tile
  const int m0   = blockIdx.x * 32;

  const float* x2s = norms;
  const float* y2s = norms + 8192;

  // Stage A panel once: 32x512 bf16
#pragma unroll
  for (int i = 0; i < 8; ++i) {
    int f   = tid + i * 256;
    int row = f >> 6;          // 0..31
    int k8  = f & 63;          // 0..63
    uint4v v = *(const uint4v*)(Xbf + (size_t)(m0 + row) * 512 + k8 * 8);
    *(uint4v*)&As[row * 520 + k8 * 8] = v;
  }

  float xr[2][8];
#pragma unroll
  for (int i = 0; i < 2; ++i)
#pragma unroll
    for (int r = 0; r < 8; ++r)
      xr[i][r] = x2s[m0 + i * 16 + r + (lane >> 4) * 8];

  float acc[2][8];
#pragma unroll
  for (int i = 0; i < 2; ++i)
#pragma unroll
    for (int r = 0; r < 8; ++r) acc[i][r] = 0.f;

  for (int nt = 0; nt < 16; ++nt) {
    const int n0 = nt * 512;
    v8f c[2][4];
#pragma unroll
    for (int i = 0; i < 2; ++i)
#pragma unroll
      for (int j = 0; j < 4; ++j) c[i][j] = (v8f){};

    __syncthreads();   // everyone done reading buffers from the previous N-tile

#if KRR_USE_TDM
    if (wid == 0)
      tdm_load_b_tile(Tbf, n0, 0, (unsigned)(size_t)&Bs[0][0]);
#else
    {
#pragma unroll
      for (int i = 0; i < 8; ++i) {
        int f = tid + i * 256, row = f >> 2, k8 = f & 3;
        uint4v v = *(const uint4v*)(Tbf + (size_t)(n0 + row) * 512 + k8 * 8);
        *(uint4v*)&Bs[0][row * 40 + k8 * 8] = v;
      }
    }
#endif

    for (int kt = 0; kt < 16; ++kt) {
      const int cur = kt % 3;
      const int nxt = (kt + 1) % 3;

#if KRR_USE_TDM
      if (wid == 0) {
        if (kt < 15) {
          tdm_load_b_tile(Tbf, n0, kt + 1, (unsigned)(size_t)&Bs[nxt][0]);
          __builtin_amdgcn_s_wait_tensorcnt(1);   // wait for Bs[cur]; kt+1's TDM may stay in flight
        } else {
          __builtin_amdgcn_s_wait_tensorcnt(0);   // last step: drain
        }
      }
      __syncthreads();   // Bs[cur] staged; prior readers of Bs[nxt] already fenced (kt-1 barrier)
#else
      uint4v stg[8];
      if (kt < 15) {
#pragma unroll
        for (int i = 0; i < 8; ++i) {
          int f = tid + i * 256, row = f >> 2, k8 = f & 3;
          stg[i] = *(const uint4v*)(Tbf + (size_t)(n0 + row) * 512 + (kt + 1) * 32 + k8 * 8);
        }
      }
      __syncthreads();
#endif

      // Compute: 2 A-frags reused across 4 B-frags -> 8 WMMAs
      v16bf a0 = load_frag_lds(&As[0 * 520 + kt * 32], 520, lane);
      v16bf a1 = load_frag_lds(&As[16 * 520 + kt * 32], 520, lane);
#pragma unroll
      for (int j = 0; j < 4; ++j) {
        v16bf b = load_frag_lds(&Bs[cur][(wid * 64 + j * 16) * 40], 40, lane);
        c[0][j] = __builtin_amdgcn_wmma_f32_16x16x32_bf16(false, a0, false, b, (short)0, c[0][j], false, false);
        c[1][j] = __builtin_amdgcn_wmma_f32_16x16x32_bf16(false, a1, false, b, (short)0, c[1][j], false, false);
      }

#if !KRR_USE_TDM
      if (kt < 15) {
#pragma unroll
        for (int i = 0; i < 8; ++i) {
          int f = tid + i * 256, row = f >> 2, k8 = f & 3;
          *(uint4v*)&Bs[nxt][row * 40 + k8 * 8] = stg[i];
        }
      }
#endif
    }

    // Fused RBF epilogue: C/D layout -> (M = i*16 + r + 8*(lane>=16), N = lane&15)
#pragma unroll
    for (int j = 0; j < 4; ++j) {
      int gn    = n0 + wid * 64 + j * 16 + (lane & 15);
      float y2v = y2s[gn];
      float dc  = dual[gn];
#pragma unroll
      for (int i = 0; i < 2; ++i) {
#pragma unroll
        for (int r = 0; r < 8; ++r) {
          float sq = xr[i][r] + y2v - 2.f * c[i][j][r];
          sq = fmaxf(sq, 0.f);
          acc[i][r] += __expf(-GAMMA * sq) * dc;
        }
      }
    }
  }

  // Reduce across 16-lane N groups, then across the 8 waves via LDS
#pragma unroll
  for (int i = 0; i < 2; ++i) {
#pragma unroll
    for (int r = 0; r < 8; ++r) {
      float v = acc[i][r];
      v += __shfl_xor(v, 1, 32);
      v += __shfl_xor(v, 2, 32);
      v += __shfl_xor(v, 4, 32);
      v += __shfl_xor(v, 8, 32);
      if ((lane & 15) == 0)
        red[wid][i * 16 + r + (lane >> 4) * 8] = v;
    }
  }
  __syncthreads();
  if (tid < 32) {
    float s = 0.f;
#pragma unroll
    for (int w = 0; w < 8; ++w) s += red[w][tid];
    out[m0 + tid] = s;
  }
}

extern "C" void kernel_launch(void* const* d_in, const int* in_sizes, int n_in,
                              void* d_out, int out_size, void* d_ws, size_t ws_size,
                              hipStream_t stream) {
  const float* X    = (const float*)d_in[0];  // (8192, 512)
  const float* T    = (const float*)d_in[1];  // (8192, 512)
  const float* dual = (const float*)d_in[2];  // (8192,)
  float* out = (float*)d_out;                 // (8192,)

  // Workspace layout: Xbf (8 MB) | Tbf (8 MB) | norms (64 KB)
  unsigned short* Xbf = (unsigned short*)d_ws;
  unsigned short* Tbf = Xbf + (size_t)8192 * 512;
  float* norms        = (float*)(Tbf + (size_t)8192 * 512);

  krr_prep<<<2048, 256, 0, stream>>>(X, T, Xbf, Tbf, norms);
  krr_main<<<256, 256, 0, stream>>>(Xbf, Tbf, dual, norms, out);
}
